// QWindowAttention_61718680043570
// MI455X (gfx1250) — matrix operand order repaired and checked by reading, not compiled, not run on hardware
//
#include <hip/hip_runtime.h>
#include <stdint.h>

typedef _Float16 f16;
typedef __attribute__((ext_vector_type(4)))  _Float16 v4h;
typedef __attribute__((ext_vector_type(8)))  _Float16 v8h;
typedef __attribute__((ext_vector_type(16))) _Float16 v16h;
typedef __attribute__((ext_vector_type(8)))  float    v8f;

#define NTOK   49
#define CDIM   128
#define HEADS  4
#define HD     32
#define NWIN   64
#define SCALE  0.17677669529663687f   // 32^-0.5

// LDS strides (in elements) padded for bank-conflict-free b128 operand fetches
#define XS  136   // x tile / o_cat: 64 rows x 128 cols (+8 pad) f16
#define QS  40    // q/k: per head 64 rows x 32 cols (+8 pad) f16
#define VS  72    // v_t: per head 32 rows(d) x 64 cols(m) (+8 pad) f16
#define ATS 72    // attn f16: per head 64 x 64 (+8) f16
#define AFS 66    // attn f32: per head 64 x 64 (+2) f32

// LDS layout (bytes); slot A is phase-aliased:
//   [x_f16 | x_f32 staging] -> attn_f32 -> o_cat
#define OFF_X32 32768                      // f32 x staging inside slot A
#define OFF_Q  67584                       // = 4*64*66*4 (attn_f32 size)
#define OFF_K  (OFF_Q + 20480)             // q size = 4*64*40*2
#define OFF_V  (OFF_K + 20480)
#define SMEMB  (OFF_V + 18432)             // v_t = 4*32*72*2  -> 126976 B total

__device__ __forceinline__ v16h make16(v8h lo, v8h hi) {
  return __builtin_shufflevector(lo, hi, 0,1,2,3,4,5,6,7,8,9,10,11,12,13,14,15);
}

// A operand (16x32 f16 tile) from row-major MxK storage, row stride S halves.
// ISA 7.12.2: lane<16 holds row M=lane, K = kb+0..7 and kb+16..23;
//             lane>=16 holds row M=lane-16, K = kb+8..15 and kb+24..31.
__device__ __forceinline__ v16h loadA(const f16* base, int S, int mb, int kb, int lane) {
  int row = mb + (lane & 15);
  int k0  = kb + ((lane >> 4) << 3);
  const f16* p = base + row * S + k0;
  return make16(*(const v8h*)p, *(const v8h*)(p + 16));
}

// B operand (32x16 f16 tile) from "B-transposed" storage Bt[n][k] (stride S):
// lane<16 holds col N=lane, K = kb+0..7 and kb+16..23; lane>=16 holds col N=lane-16, K = kb+8..15 and kb+24..31.
__device__ __forceinline__ v16h loadB(const f16* base, int S, int nb, int kb, int lane) {
  int row = nb + (lane & 15);
  int k0  = kb + ((lane >> 4) << 3);
  const f16* p = base + row * S + k0;
  return make16(*(const v8h*)p, *(const v8h*)(p + 16));
}

__device__ __forceinline__ v8f wmma(v16h a, v16h b, v8f c) {
  return __builtin_amdgcn_wmma_f32_16x16x32_f16(false, a, false, b, (short)0, c, false, false);
}

// ---------------- pre-pass kernels ----------------
__global__ void prep_weights(const float* __restrict__ qkv_w, const float* __restrict__ proj_w,
                             f16* __restrict__ wq, f16* __restrict__ wp) {
  int i = blockIdx.x * 256 + threadIdx.x;
  if (i < 3 * CDIM * CDIM) wq[i] = (f16)qkv_w[i];
  if (i < CDIM * CDIM)     wp[i] = (f16)proj_w[i];
}

__global__ void prep_bias(const float* __restrict__ rpb, const int* __restrict__ rel,
                          float* __restrict__ bg) {
  int i = blockIdx.x * 256 + threadIdx.x;
  if (i < HEADS * NTOK * NTOK) {
    int h = i / (NTOK * NTOK);
    int r = i - h * (NTOK * NTOK);
    bg[i] = rpb[rel[r] * HEADS + h];
  }
}

// ---------------- main kernel: one block per window ----------------
__global__ void __launch_bounds__(256)
win_attn(const float* __restrict__ x, const float* __restrict__ mask,
         const float* __restrict__ qkv_b, const float* __restrict__ proj_b,
         const f16* __restrict__ wq, const f16* __restrict__ wp,
         const float* __restrict__ bg, float* __restrict__ out) {
  __shared__ __align__(16) unsigned char smem[SMEMB];
  f16*   xs   = (f16*)smem;                 // 64 x XS f16 (phase 1-2)
  float* xs32 = (float*)(smem + OFF_X32);   // raw 49x128 f32 (async staging)
  float* af   = (float*)smem;               // 4 x 64 x AFS f32 (phase 2.5-4)
  f16*   oc   = (f16*)smem;                 // 64 x XS f16 (phase 5-6)
  f16*   qs   = (f16*)(smem + OFF_Q);
  f16*   ks   = (f16*)(smem + OFF_K);
  f16*   vt   = (f16*)(smem + OFF_V);
  f16*   at16 = (f16*)(smem + OFF_Q);       // aliases q/k after QK^T

  const int tid  = threadIdx.x;
  const int lane = tid & 31;
  const int wv   = tid >> 5;
  const int blk  = blockIdx.x;

  // ---- phase 0: async-copy raw x tile (f32) to LDS (ASYNCcnt path) ----
  const float* xb = x + (size_t)blk * NTOK * CDIM;
  {
    unsigned ldsbase = (unsigned)(uintptr_t)xs32;
    for (int i = tid; i < (NTOK * CDIM) / 4; i += 256) {   // 1568 x b128
      unsigned lds_addr = ldsbase + (unsigned)i * 16u;
      unsigned goff     = (unsigned)i * 16u;
      asm volatile("global_load_async_to_lds_b128 %0, %1, %2"
                   :: "v"(lds_addr), "v"(goff), "s"(xb) : "memory");
    }
    asm volatile("s_wait_asynccnt 0" ::: "memory");
  }
  __syncthreads();

  // ---- phase 1: convert to f16 tile (pad rows 49..63 with zeros) ----
  for (int i = tid; i < 64 * 32; i += 256) {
    int row = i >> 5, c4 = (i & 31) << 2;
    float4 v = make_float4(0.f, 0.f, 0.f, 0.f);
    if (row < NTOK) v = *(const float4*)(xs32 + row * CDIM + c4);
    v4h h; h[0] = (f16)v.x; h[1] = (f16)v.y; h[2] = (f16)v.z; h[3] = (f16)v.w;
    *(v4h*)(xs + row * XS + c4) = h;
  }
  __syncthreads();

  // ---- phase 2: QKV GEMM  (64x128) @ (128x384): 96 tiles, 12 per wave ----
  for (int i = 0; i < 12; ++i) {
    int ti = wv + (i << 3);
    int mt = ti & 3, nt = ti >> 2;
    int mb = mt << 4, nb = nt << 4;
    v8f acc = {};
    for (int kb = 0; kb < CDIM; kb += 32) {
      v16h a = loadA(xs, XS, mb, kb, lane);
      v16h b = loadB(wq, CDIM, nb, kb, lane);   // wq row-major [384][128] == B^T
      acc = wmma(a, b, acc);
    }
    int which = nt >> 3;            // 0=q, 1=k, 2=v (uniform per tile)
    int head  = (nt >> 1) & 3;
    int d     = ((nt & 1) << 4) + (lane & 15);
    float bias = qkv_b[nb + (lane & 15)];
    int rbase = mb + ((lane >> 4) << 3);
    if (which == 0) {
      f16* p = qs + (head * 64) * QS + d;
      #pragma unroll
      for (int r = 0; r < 8; ++r) p[(rbase + r) * QS] = (f16)((acc[r] + bias) * SCALE);
    } else if (which == 1) {
      f16* p = ks + (head * 64) * QS + d;
      #pragma unroll
      for (int r = 0; r < 8; ++r) p[(rbase + r) * QS] = (f16)(acc[r] + bias);
    } else {
      f16* p = vt + (head * 32 + d) * VS;       // transposed: [head][d][m]
      #pragma unroll
      for (int r = 0; r < 8; ++r) p[rbase + r] = (f16)(acc[r] + bias);
    }
  }
  __syncthreads();

  // ---- phase 2.5: prefill attn_f32 with bias + window mask (coalesced) ----
  {
    const float* mrow = mask + (size_t)(blk & (NWIN - 1)) * NTOK * NTOK;
    for (int i = tid; i < HEADS * NTOK * NTOK; i += 256) {
      int h = i / (NTOK * NTOK);
      int r = i - h * (NTOK * NTOK);
      int n = r / NTOK;
      int m = r - n * NTOK;
      af[(h * 64 + n) * AFS + m] = bg[i] + mrow[r];
    }
  }
  __syncthreads();

  // ---- phase 3: QK^T per head, accumulating onto bias+mask in LDS ----
  for (int i = 0; i < 8; ++i) {
    int t = (wv << 3) + i;
    int head = t >> 4, rem = t & 15, at = rem >> 2, bt = rem & 3;
    v16h a = loadA(qs + head * 64 * QS, QS, at << 4, 0, lane);
    v16h b = loadB(ks + head * 64 * QS, QS, bt << 4, 0, lane);  // ks[m][d] == B^T
    v8f acc = {};
    acc = wmma(a, b, acc);
    float* dst = af + (head * 64 + (at << 4) + ((lane >> 4) << 3)) * AFS
                    + (bt << 4) + (lane & 15);
    #pragma unroll
    for (int r = 0; r < 8; ++r) dst[r * AFS] += acc[r];
  }
  __syncthreads();

  // ---- phase 4: softmax (one thread per (head,row); LDS only) ----
  if (tid < HEADS * NTOK) {
    int h = tid / NTOK;
    int n = tid - h * NTOK;
    float* L = af + (h * 64 + n) * AFS;
    float mx = -1e30f;
    for (int m = 0; m < NTOK; ++m) mx = fmaxf(mx, L[m]);
    float s = 0.f;
    for (int m = 0; m < NTOK; ++m) { float e = __expf(L[m] - mx); L[m] = e; s += e; }
    float inv = 1.0f / s;
    f16* arow = at16 + (h * 64 + n) * ATS;
    for (int m = 0; m < NTOK; ++m) arow[m] = (f16)(L[m] * inv);
    for (int m = NTOK; m < 64; ++m) arow[m] = (f16)0.f;   // zero K-padding
  }
  __syncthreads();

  // ---- phase 5: attn @ V per head (32 tiles, 4/wave, K=64 in 2 steps) ----
  for (int i = 0; i < 4; ++i) {
    int t = (wv << 2) + i;
    int head = t >> 3, rem = t & 7, mt = rem >> 1, dt = rem & 1;
    v8f acc = {};
    for (int kb = 0; kb < 64; kb += 32) {
      v16h a = loadA(at16 + head * 64 * ATS, ATS, mt << 4, kb, lane);
      v16h b = loadB(vt + head * 32 * VS, VS, dt << 4, kb, lane); // vt[d][m] == B^T
      acc = wmma(a, b, acc);
    }
    int col = (head << 5) + (dt << 4) + (lane & 15);
    int rb  = (mt << 4) + ((lane >> 4) << 3);
    #pragma unroll
    for (int r = 0; r < 8; ++r) oc[(rb + r) * XS + col] = (f16)acc[r];
  }
  __syncthreads();

  // ---- phase 6: proj GEMM (64x128)@(128x128), bias, fp32 store ----
  float* ob = out + (size_t)blk * NTOK * CDIM;
  for (int i = 0; i < 4; ++i) {
    int t = (wv << 2) + i;
    int mt = t >> 3, nt = t & 7;
    v8f acc = {};
    for (int kb = 0; kb < CDIM; kb += 32) {
      v16h a = loadA(oc, XS, mt << 4, kb, lane);
      v16h b = loadB(wp, CDIM, nt << 4, kb, lane);   // wp row-major [128][128] == B^T
      acc = wmma(a, b, acc);
    }
    int col = (nt << 4) + (lane & 15);
    float pb = proj_b[col];
    int rb = (mt << 4) + ((lane >> 4) << 3);
    #pragma unroll
    for (int r = 0; r < 8; ++r) {
      int row = rb + r;
      if (row < NTOK) ob[row * CDIM + col] = acc[r] + pb;
    }
  }
}

// ---------------- launch ----------------
extern "C" void kernel_launch(void* const* d_in, const int* in_sizes, int n_in,
                              void* d_out, int out_size, void* d_ws, size_t ws_size,
                              hipStream_t stream) {
  const float* x      = (const float*)d_in[0];
  const float* mask   = (const float*)d_in[1];
  const float* qkv_w  = (const float*)d_in[2];
  const float* qkv_b  = (const float*)d_in[3];
  const float* proj_w = (const float*)d_in[4];
  const float* proj_b = (const float*)d_in[5];
  const float* rpb    = (const float*)d_in[6];
  const int*   rel    = (const int*)d_in[7];
  float* out = (float*)d_out;

  char* ws = (char*)d_ws;
  f16*   wq = (f16*)(ws);                  // 384*128 f16  = 98304 B
  f16*   wp = (f16*)(ws + 98304);          // 128*128 f16  = 32768 B
  float* bg = (float*)(ws + 131072);       // 4*49*49 f32  = 38416 B

  prep_weights<<<192, 256, 0, stream>>>(qkv_w, proj_w, wq, wp);
  prep_bias<<<38, 256, 0, stream>>>(rpb, rel, bg);
  win_attn<<<4096, 256, 0, stream>>>(x, mask, qkv_b, proj_b, wq, wp, bg, out);
}
